// CosineSimilarity_56435870269927
// MI455X (gfx1250) — matrix operand order repaired and checked by reading, not compiled, not run on hardware
//
#include <hip/hip_runtime.h>
#include <hip/hip_bf16.h>
#include <cstdint>

// ---------------------------------------------------------------------------
// Pairwise cosine similarity on MI455X (gfx1250, wave32):
//   C[n,m] = <A_n, B_m> / (max(|A_n|,eps) * max(|B_m|,eps))
// fp32 row-normalize -> bf16 copies in workspace -> WMMA bf16 GEMM
// (v_wmma_f32_16x16x32_bf16) with triple-buffered ASYNC global->LDS staging
// (global_load_async_to_lds_b128, partial s_wait_asynccnt).
// ---------------------------------------------------------------------------

typedef __bf16 bf16x16 __attribute__((ext_vector_type(16)));
typedef __bf16 bf16x4  __attribute__((ext_vector_type(4)));
typedef float  floatx8 __attribute__((ext_vector_type(8)));

union FragU {
    bf16x16 v;
    uint4   q[2];
};

#define COS_EPS 1e-8f
#define D_COLS  1024
#define LDSW    40   // bf16 elements per LDS row: 32 data + 8 pad (80B stride)
#define NBUF    3    // triple-buffered K pipeline

// --- CDNA5 async global->LDS copy (tracked by ASYNCcnt, no VGPR data) ------
__device__ __forceinline__ unsigned lds_off(const void* p) {
    // flat LDS pointers carry the LDS byte offset in the low 32 bits
    return (unsigned)(uintptr_t)p;
}
__device__ __forceinline__ void async_copy_b128(unsigned ldst, const void* gsrc,
                                                int ioff) {
    asm volatile("global_load_async_to_lds_b128 %0, %1, off offset:%2"
                 :: "v"(ldst), "v"(gsrc), "i"(ioff)
                 : "memory");
}
__device__ __forceinline__ void wait_async_le4() {
    asm volatile("s_wait_asynccnt 4" ::: "memory");
}
__device__ __forceinline__ void wait_async_0() {
    asm volatile("s_wait_asynccnt 0" ::: "memory");
}

// ---------------------------------------------------------------------------
// Kernel 1: per-row L2 norm + normalize + fp32 -> bf16 conversion.
// ---------------------------------------------------------------------------
__global__ __launch_bounds__(256) void rownorm_bf16_kernel(
    const float* __restrict__ in, __bf16* __restrict__ out) {
    const int row = blockIdx.x;
    const int t   = threadIdx.x;
    const size_t base = (size_t)row * D_COLS;

    float4 v = ((const float4*)(in + base))[t];
    float  ss = v.x * v.x + v.y * v.y + v.z * v.z + v.w * v.w;

    __shared__ float red[256];
    red[t] = ss;
    __syncthreads();
#pragma unroll
    for (int s = 128; s > 0; s >>= 1) {
        if (t < s) red[t] += red[t + s];
        __syncthreads();
    }
    if (t == 0) red[0] = 1.0f / fmaxf(sqrtf(red[0]), COS_EPS);
    __syncthreads();
    const float scale = red[0];

    bf16x4 o;
    o[0] = (__bf16)(v.x * scale);
    o[1] = (__bf16)(v.y * scale);
    o[2] = (__bf16)(v.z * scale);
    o[3] = (__bf16)(v.w * scale);
    ((bf16x4*)(out + base))[t] = o;
}

// ---------------------------------------------------------------------------
// Kernel 2: C = A * B^T. 256 threads = 8 waves; 128x128 tile; waves 2(M)x4(N);
// each wave computes 4x2 WMMA tiles. K step = 32, 3-stage async pipeline:
//   iter kt:  wait G(kt) (asynccnt<=4, in-order completion) -> barrier ->
//             issue G(kt+2) -> ds_load fragments -> 8x wmma.
// ---------------------------------------------------------------------------
template <int NCOLS, int K>
__global__ __launch_bounds__(256) void cosine_wmma_gemm(
    const __bf16* __restrict__ A, const __bf16* __restrict__ B,
    float* __restrict__ C) {

    __shared__ __bf16 sA[NBUF][128 * LDSW];
    __shared__ __bf16 sB[NBUF][128 * LDSW];

    const int tid  = threadIdx.x;
    const int lane = tid & 31;
    const int wave = tid >> 5;
    const int wm   = wave & 1;   // 0..1 -> 64-row group
    const int wn   = wave >> 1;  // 0..3 -> 32-col group

    const size_t arow0 = (size_t)blockIdx.y * 128;
    const size_t brow0 = (size_t)blockIdx.x * 128;

    // staging: 128 rows x 64B per matrix per K-step; 4 async b128 per thread
    const int r0 = tid >> 2;        // rows 0..63
    const int r1 = r0 + 64;         // rows 64..127
    const int sg = (tid & 3) * 8;   // bf16 offset of 16B segment within K-chunk

    // fragment addressing (ISA 7.12.2 layouts)
    const int fr = lane & 15;       // row (A) / column (B) within 16
    const int fh = lane >> 4;       // half select

    floatx8 acc[4][2] = {};

    // loop-invariant global bases; per-iteration K advance via imm offset
    const __bf16* gA0 = A + (arow0 + r0) * K + sg;
    const __bf16* gA1 = A + (arow0 + r1) * K + sg;
    const __bf16* gB0 = B + (brow0 + r0) * K + sg;
    const __bf16* gB1 = B + (brow0 + r1) * K + sg;

    unsigned lA0[NBUF], lA1[NBUF], lB0[NBUF], lB1[NBUF];
#pragma unroll
    for (int b = 0; b < NBUF; ++b) {
        lA0[b] = lds_off(&sA[b][r0 * LDSW + sg]);
        lA1[b] = lds_off(&sA[b][r1 * LDSW + sg]);
        lB0[b] = lds_off(&sB[b][r0 * LDSW + sg]);
        lB1[b] = lds_off(&sB[b][r1 * LDSW + sg]);
    }

    auto stage = [&](int buf, int kt) {
        const int ko = kt * 64;  // bytes: 32 bf16 per K-step
        async_copy_b128(lA0[buf], gA0, ko);
        async_copy_b128(lA1[buf], gA1, ko);
        async_copy_b128(lB0[buf], gB0, ko);
        async_copy_b128(lB1[buf], gB1, ko);
    };

    constexpr int KT = K / 32;
    static_assert(KT >= 2, "pipeline needs >= 2 K-steps");
    stage(0, 0);
    stage(1, 1);

#pragma unroll
    for (int kt = 0; kt < KT; ++kt) {
        const int cur = kt % NBUF;

        if (kt < KT - 1) wait_async_le4();  // retire G(kt); keep G(kt+1) in flight
        else             wait_async_0();    // last group
        __syncthreads();                    // all waves' G(kt) visible in LDS

        if (kt + 2 < KT) stage((kt + 2) % NBUF, kt + 2);

        // A fragment (16x32): lane L holds row L%16, K in {h*8..+7, 16+h*8..+7}
        FragU fa[4];
#pragma unroll
        for (int mi = 0; mi < 4; ++mi) {
            const int ar = wm * 64 + mi * 16 + fr;
            fa[mi].q[0] = *(const uint4*)&sA[cur][ar * LDSW + fh * 8];
            fa[mi].q[1] = *(const uint4*)&sA[cur][ar * LDSW + 16 + fh * 8];
        }
        // B fragment (32x16 = B^T tile): lane L holds column L%16, K in {h*8..+7, 16+h*8..+7}
        FragU fb[2];
#pragma unroll
        for (int ni = 0; ni < 2; ++ni) {
            const int br = wn * 32 + ni * 16 + fr;
            fb[ni].q[0] = *(const uint4*)&sB[cur][br * LDSW + fh * 8];
            fb[ni].q[1] = *(const uint4*)&sB[cur][br * LDSW + 16 + fh * 8];
        }

#pragma unroll
        for (int mi = 0; mi < 4; ++mi)
#pragma unroll
            for (int ni = 0; ni < 2; ++ni)
                acc[mi][ni] = __builtin_amdgcn_wmma_f32_16x16x32_bf16(
                    false, fa[mi].v, false, fb[ni].v,
                    (short)0, acc[mi][ni], false, false);
    }

    // Epilogue: C/D layout — lanes 0..15: m = vgpr, n = lane; lanes 16..31: m = 8+vgpr
#pragma unroll
    for (int mi = 0; mi < 4; ++mi) {
#pragma unroll
        for (int ni = 0; ni < 2; ++ni) {
            const size_t row = arow0 + wm * 64 + mi * 16 + fh * 8;
            const size_t col = brow0 + wn * 32 + ni * 16 + fr;
            float* cp = C + row * (size_t)NCOLS + col;
#pragma unroll
            for (int v = 0; v < 8; ++v)
                cp[(size_t)v * NCOLS] = acc[mi][ni][v];  // imm-offset stores
        }
    }
}

// ---------------------------------------------------------------------------
extern "C" void kernel_launch(void* const* d_in, const int* in_sizes, int n_in,
                              void* d_out, int out_size, void* d_ws, size_t ws_size,
                              hipStream_t stream) {
    const float* in1 = (const float*)d_in[0];
    const float* in2 = (const float*)d_in[1];
    const int D = D_COLS;
    const int N = in_sizes[0] / D;  // 8192 rows of input1
    const int M = in_sizes[1] / D;  // 8192 rows of input2

    __bf16* Abf = (__bf16*)d_ws;                 // N*D bf16
    __bf16* Bbf = Abf + (size_t)N * D;           // M*D bf16 (total 32 MB)

    rownorm_bf16_kernel<<<N, 256, 0, stream>>>(in1, Abf);
    rownorm_bf16_kernel<<<M, 256, 0, stream>>>(in2, Bbf);

    dim3 grid(M / 128, N / 128);
    cosine_wmma_gemm<8192, D_COLS><<<grid, 256, 0, stream>>>(Abf, Bbf, (float*)d_out);
}
